// Rotanet_22754736734581
// MI455X (gfx1250) — compile-verified, matrix operand order checked by reading, not run on hardware
//
#include <hip/hip_runtime.h>
#include <hip/hip_bf16.h>

// ---------------------------------------------------------------------------
// Sudoku RRN forward on gfx1250 (CDNA5, wave32, WMMA + async LDS copy).
// Activations live in memory as bf16 (WMMA-native); accumulation, LSTM cell
// state and gate pre-activations stay f32. All GEMMs use
// v_wmma_f32_16x16x32_bf16; weights are pre-packed into LDS-image layout and
// staged per block with global_load_async_to_lds_b128 (ASYNCcnt).
// ---------------------------------------------------------------------------

typedef __attribute__((ext_vector_type(16))) __bf16 v16bf;
typedef __attribute__((ext_vector_type(8)))  float  v8f;

#define kB 128
#define kN 81
#define kD 9
#define kH 96
#define kM (kB * kN * kD)      // 93312 rows
#define kRT (kM / 16)          // 5832 row tiles of 16 (= 729 blocks x 8 waves)
#define MAXNB 24
#define SLAB (96 * 104)        // u16 elements per packed weight slab (LDS image)

typedef unsigned short u16;
typedef unsigned int   u32;

__device__ __forceinline__ u16 f2bf(float f) {        // native cvt (RNE)
  __bf16 h = (__bf16)f;
  return __builtin_bit_cast(u16, h);
}
__device__ __forceinline__ float bf2f(u16 s) {
  return __uint_as_float((u32)s << 16);
}

union Frag { v16bf v; u32 u[8]; };

// ---------------------------------------------------------------------------
// GEMM: C[M, Nout] = act( sum_s A_s[M,96](bf16) @ W_s + bias )
//   Weight image: NSRC slabs of [96][104] bf16 already in Wt[n][k] layout;
//   block copies it to LDS with async b128 copies (ASYNCcnt), then runs
//   16x16x32 bf16 WMMA. 256 threads = 8 waves; wave w owns row tile
//   blockIdx.x*8+w; blockIdx.y selects the 96-column group. Per k-step all
//   six B fragments are loaded first (one dscnt wait), then 6 WMMAs issue
//   back-to-back.
// ---------------------------------------------------------------------------
template <int NSRC, int ACT, int OUTF, int OUTB>
__global__ __launch_bounds__(256) void gemm96_k(
    const u16* __restrict__ A0, const u16* __restrict__ A1,
    const u16* __restrict__ A2, const u16* __restrict__ Wimg,
    const float* __restrict__ bias, float* __restrict__ Cf,
    u16* __restrict__ Cb, int Nout) {
  __shared__ __align__(16) u16 ldsW[NSRC][96][104];   // pad 104 vs 64 banks
  __shared__ float ldsB[96];

  const int tid  = threadIdx.x;
  const int nofs = blockIdx.y * 96;

  // Async bulk copy of the pre-packed weight image into LDS.
  {
    const u16* src = Wimg + (size_t)blockIdx.y * (NSRC * SLAB);
    unsigned ldsBase = (unsigned)(uintptr_t)(void*)&ldsW[0][0][0];  // addr[31:0]
    constexpr int SEGS = NSRC * SLAB * 2 / 16;        // 16-byte segments
    for (int seg = tid; seg < SEGS; seg += 256) {
      unsigned lds = ldsBase + seg * 16;
      unsigned long long ga = (unsigned long long)(uintptr_t)src + (unsigned)seg * 16u;
      asm volatile("global_load_async_to_lds_b128 %0, %1, off"
                   :: "v"(lds), "v"(ga) : "memory");
    }
    if (tid < 96) ldsB[tid] = bias ? bias[nofs + tid] : 0.0f;
    asm volatile("s_wait_asynccnt 0x0" ::: "memory");
  }
  __syncthreads();

  const int wave = tid >> 5;
  const int lane = tid & 31;
  const int mrow = lane & 15;
  const int kb   = (lane & 16) ? 8 : 0;   // 16-bit frag: lanes>=16 hold K base 8
  const int rt   = blockIdx.x * 8 + wave; // exactly kRT tiles, no loop

  v8f zero = {0.f, 0.f, 0.f, 0.f, 0.f, 0.f, 0.f, 0.f};
  v8f acc[6] = {zero, zero, zero, zero, zero, zero};

#pragma unroll
  for (int s = 0; s < NSRC; ++s) {
    const u16* A = (s == 0) ? A0 : ((s == 1) ? A1 : A2);
    const u16* arow = A + (size_t)(rt * 16 + mrow) * 96;
#pragma unroll
    for (int kt = 0; kt < 3; ++kt) {                 // K = 96 = 3 x 32
      // A fragment: bf16 already in WMMA A layout order -> two b128 loads.
      uint4 a0 = *(const uint4*)(arow + kt * 32 + kb);        // K = kb+0..7
      uint4 a1 = *(const uint4*)(arow + kt * 32 + kb + 16);   // K = kb+16..23
      Frag af;
      af.u[0] = a0.x; af.u[1] = a0.y; af.u[2] = a0.z; af.u[3] = a0.w;
      af.u[4] = a1.x; af.u[5] = a1.y; af.u[6] = a1.z; af.u[7] = a1.w;
      // Batch-load all six B fragments for this k-step (one dscnt wait),
      // then issue the six WMMAs back-to-back.
      Frag bfm[6];
#pragma unroll
      for (int nt = 0; nt < 6; ++nt) {
        const u16* wp = &ldsW[s][nt * 16 + mrow][kt * 32 + kb];
        uint4 q0 = *(const uint4*)(wp);              // K = kb+0..7
        uint4 q1 = *(const uint4*)(wp + 16);         // K = kb+16..23
        bfm[nt].u[0] = q0.x; bfm[nt].u[1] = q0.y;
        bfm[nt].u[2] = q0.z; bfm[nt].u[3] = q0.w;
        bfm[nt].u[4] = q1.x; bfm[nt].u[5] = q1.y;
        bfm[nt].u[6] = q1.z; bfm[nt].u[7] = q1.w;
      }
#pragma unroll
      for (int nt = 0; nt < 6; ++nt) {
        acc[nt] = __builtin_amdgcn_wmma_f32_16x16x32_bf16(
            false, af.v, false, bfm[nt].v, (short)0, acc[nt], false, false);
      }
    }
  }

  // Epilogue: C/D layout -> vgpr i, lanes<16: M=i, lanes>=16: M=8+i; N=lane%16.
  const int col   = lane & 15;
  const int rbase = rt * 16 + ((lane >> 4) << 3);
#pragma unroll
  for (int nt = 0; nt < 6; ++nt) {
    float bv = ldsB[nt * 16 + col];
#pragma unroll
    for (int i = 0; i < 8; ++i) {
      float v = acc[nt][i] + bv;
      if constexpr (ACT) v = fmaxf(v, 0.0f);
      size_t idx = (size_t)(rbase + i) * Nout + nofs + nt * 16 + col;
      if constexpr (OUTF) Cf[idx] = v;
      if constexpr (OUTB) Cb[idx] = f2bf(v);
    }
  }
}

// ---------------------------------------------------------------------------
// Weight prep: pack one [96 x 96] weight (transposed if tr) into a bf16 slab
// [n=96][k=104] identical to the GEMM's LDS image. Pad columns 96..103 = 0.
// ---------------------------------------------------------------------------
__global__ void prep_w_k(const float* __restrict__ W, int tr, int nofs,
                         u16* __restrict__ dst) {
  int t = blockIdx.x * blockDim.x + threadIdx.x;
  if (t >= SLAB) return;
  int n = t / 104, k = t % 104;
  float v = 0.0f;
  if (k < 96) v = tr ? W[(size_t)k * 96 + (nofs + n)]
                     : W[(size_t)(nofs + n) * 96 + k];
  dst[t] = f2bf(v);
}

// ---------------------------------------------------------------------------
// Encoder: X = relu(mask @ enc_W1 + b1) @ enc_W2 + b2 -> bf16.
// ---------------------------------------------------------------------------
__global__ void encode_k(const int* __restrict__ inputs,
                         const float* __restrict__ W1, const float* __restrict__ b1,
                         const float* __restrict__ W2, const float* __restrict__ b2,
                         u16* __restrict__ X) {
  int t = blockIdx.x * blockDim.x + threadIdx.x;
  if (t >= kM * kH) return;
  int j = t % kH, m = t / kH;
  int d = m % kD, bn = m / kD;
  int inp = inputs[bn];
  float m0 = (inp == 0) ? 1.f : 0.f;                  // blank
  float m1 = (inp == d + 1) ? 1.f : 0.f;              // equals digit
  float m2 = (inp != 0 && inp != d + 1) ? 1.f : 0.f;  // filled with other
  float x = b2[j];
#pragma unroll
  for (int e = 0; e < 16; ++e) {
    float hsum = fmaf(m0, W1[e], fmaf(m1, W1[16 + e], fmaf(m2, W1[32 + e], b1[e])));
    x = fmaf(fmaxf(hsum, 0.f), W2[e * kH + j], x);
  }
  X[t] = f2bf(x);
}

// ---------------------------------------------------------------------------
// msg_cell[b,n,e,h] = sum_d in[b,n,d,h] * adj_cell[d,e]   (bf16 in/out, f32 acc)
// ---------------------------------------------------------------------------
__global__ void msg_cell_k(const u16* __restrict__ in,
                           const float* __restrict__ adjc,
                           u16* __restrict__ out) {
  int t = blockIdx.x * blockDim.x + threadIdx.x;
  const int TOT = kB * kN * (kH / 4);
  if (t >= TOT) return;
  int h4 = t % (kH / 4), bn = t / (kH / 4);
  const u16* base = in + (size_t)bn * kD * kH + h4 * 4;
  float acc[kD][4];
#pragma unroll
  for (int d = 0; d < kD; ++d)
#pragma unroll
    for (int q = 0; q < 4; ++q) acc[d][q] = 0.f;
  for (int e = 0; e < kD; ++e) {
    uint2 p = *(const uint2*)(base + (size_t)e * kH);
    float v0 = bf2f((u16)(p.x & 0xffff)), v1 = bf2f((u16)(p.x >> 16));
    float v2 = bf2f((u16)(p.y & 0xffff)), v3 = bf2f((u16)(p.y >> 16));
#pragma unroll
    for (int d = 0; d < kD; ++d) {
      float w = adjc[e * kD + d];
      acc[d][0] = fmaf(v0, w, acc[d][0]); acc[d][1] = fmaf(v1, w, acc[d][1]);
      acc[d][2] = fmaf(v2, w, acc[d][2]); acc[d][3] = fmaf(v3, w, acc[d][3]);
    }
  }
  u16* ob = out + (size_t)bn * kD * kH + h4 * 4;
  for (int d = 0; d < kD; ++d) {
    uint2 p;
    p.x = (u32)f2bf(acc[d][0]) | ((u32)f2bf(acc[d][1]) << 16);
    p.y = (u32)f2bf(acc[d][2]) | ((u32)f2bf(acc[d][3]) << 16);
    *(uint2*)(ob + (size_t)d * kH) = p;
  }
}

// ---------------------------------------------------------------------------
// Neighbor list from adj_digit (20 ones per column for Sudoku).
// ---------------------------------------------------------------------------
__global__ void build_nbr_k(const float* __restrict__ adj,
                            int* __restrict__ nbr, int* __restrict__ cnt) {
  int m = blockIdx.x * blockDim.x + threadIdx.x;
  if (m >= kN) return;
  int c = 0;
  for (int n = 0; n < kN; ++n) {
    if (adj[(size_t)n * kN + m] != 0.0f && c < MAXNB) nbr[m * MAXNB + c++] = n;
  }
  cnt[m] = c;
}

// msg_digit[b,m,d,h] = sum_{n in nbr(m)} in[b,n,d,h]   (adj values are 1.0)
__global__ void msg_digit_k(const u16* __restrict__ in,
                            const int* __restrict__ nbr, const int* __restrict__ cnt,
                            u16* __restrict__ out) {
  int t = blockIdx.x * blockDim.x + threadIdx.x;
  const int TOT = kB * kN * kD * (kH / 4);
  if (t >= TOT) return;
  int h4 = t % (kH / 4), r = t / (kH / 4);
  int d = r % kD, bm = r / kD;
  int m = bm % kN, b = bm / kN;
  float a0 = 0.f, a1 = 0.f, a2 = 0.f, a3 = 0.f;
  int ct = cnt[m];
  for (int i = 0; i < ct; ++i) {
    int n = nbr[m * MAXNB + i];
    uint2 p = *(const uint2*)(in + (((size_t)b * kN + n) * kD + d) * kH + h4 * 4);
    a0 += bf2f((u16)(p.x & 0xffff)); a1 += bf2f((u16)(p.x >> 16));
    a2 += bf2f((u16)(p.y & 0xffff)); a3 += bf2f((u16)(p.y >> 16));
  }
  uint2 p;
  p.x = (u32)f2bf(a0) | ((u32)f2bf(a1) << 16);
  p.y = (u32)f2bf(a2) | ((u32)f2bf(a3) << 16);
  *(uint2*)(out + ((size_t)bm * kD + d) * kH + h4 * 4) = p;
}

// ---------------------------------------------------------------------------
// LSTM elementwise: gates [M,384] f32, i,f,g,o order; c stays f32, h -> bf16.
// ---------------------------------------------------------------------------
__global__ void lstm_k(const float* __restrict__ gates,
                       const float* __restrict__ bih, const float* __restrict__ bhh,
                       u16* __restrict__ h, float* __restrict__ c) {
  int t = blockIdx.x * blockDim.x + threadIdx.x;
  if (t >= kM * kH) return;
  int j = t % kH, m = t / kH;
  const float* gr = gates + (size_t)m * 384;
  float iv = gr[j]        + bih[j]        + bhh[j];
  float fv = gr[kH + j]   + bih[kH + j]   + bhh[kH + j];
  float gv = gr[2*kH + j] + bih[2*kH + j] + bhh[2*kH + j];
  float ov = gr[3*kH + j] + bih[3*kH + j] + bhh[3*kH + j];
  float si = 1.0f / (1.0f + __expf(-iv));
  float sf = 1.0f / (1.0f + __expf(-fv));
  float so = 1.0f / (1.0f + __expf(-ov));
  float cn = fmaf(sf, c[t], si * tanhf(gv));
  c[t] = cn;
  h[t] = f2bf(so * tanhf(cn));
}

// Final head projection: out[m] = a[m,:] . r_W3 + r_b3  (bf16 a, f32 out)
__global__ void head_k(const u16* __restrict__ A, const float* __restrict__ w,
                       const float* __restrict__ b, float* __restrict__ out) {
  int m = blockIdx.x * blockDim.x + threadIdx.x;
  if (m >= kM) return;
  const u16* a = A + (size_t)m * kH;
  float acc = b[0];
#pragma unroll
  for (int k = 0; k < kH; k += 8) {
    uint4 q = *(const uint4*)(a + k);
    acc = fmaf(bf2f((u16)(q.x & 0xffff)), w[k + 0],
          fmaf(bf2f((u16)(q.x >> 16)),    w[k + 1],
          fmaf(bf2f((u16)(q.y & 0xffff)), w[k + 2],
          fmaf(bf2f((u16)(q.y >> 16)),    w[k + 3], acc))));
    acc = fmaf(bf2f((u16)(q.z & 0xffff)), w[k + 4],
          fmaf(bf2f((u16)(q.z >> 16)),    w[k + 5],
          fmaf(bf2f((u16)(q.w & 0xffff)), w[k + 6],
          fmaf(bf2f((u16)(q.w >> 16)),    w[k + 7], acc))));
  }
  out[m] = acc;
}

// ---------------------------------------------------------------------------
extern "C" void kernel_launch(void* const* d_in, const int* in_sizes, int n_in,
                              void* d_out, int out_size, void* d_ws, size_t ws_size,
                              hipStream_t stream) {
  (void)in_sizes; (void)n_in; (void)ws_size;
  const int*   inputs    = (const int*)d_in[0];
  const float* c0        = (const float*)d_in[1];
  const float* adj_cell  = (const float*)d_in[2];
  const float* adj_digit = (const float*)d_in[3];
  const float* enc_W1    = (const float*)d_in[4];
  const float* enc_b1    = (const float*)d_in[5];
  const float* enc_W2    = (const float*)d_in[6];
  const float* enc_b2    = (const float*)d_in[7];
  const float* cr_W      = (const float*)d_in[8];
  const float* cr_b      = (const float*)d_in[9];
  const float* dr_W      = (const float*)d_in[10];
  const float* dr_b      = (const float*)d_in[11];
  const float* gm_W1     = (const float*)d_in[12];
  const float* gm_b1     = (const float*)d_in[13];
  const float* gm_W2     = (const float*)d_in[14];
  const float* gm_b2     = (const float*)d_in[15];
  const float* Wih       = (const float*)d_in[16];
  const float* Whh       = (const float*)d_in[17];
  const float* bih       = (const float*)d_in[18];
  const float* bhh       = (const float*)d_in[19];
  const float* r_W12     = (const float*)d_in[20];
  const float* r_b12     = (const float*)d_in[21];
  const float* r_W3      = (const float*)d_in[22];
  const float* r_b3      = (const float*)d_in[23];
  float* out = (float*)d_out;
  const int iters = out_size / kM;          // [iters, B, N, D]

  // Workspace: 5 bf16 activation slots + f32 c + f32 gates + weight images.
  const size_t ELT = (size_t)kM * kH;       // elements per [M,96] slot
  char* W = (char*)d_ws;
  u16* X  = (u16*)(W);
  u16* h  = X + ELT;
  u16* u0 = h + ELT;
  u16* u1 = u0 + ELT;
  u16* u2 = u1 + ELT;
  float* c     = (float*)(W + 5 * ELT * 2);
  float* gates = (float*)(W + 5 * ELT * 2 + ELT * 4);
  u16* wimg = (u16*)(W + 5 * ELT * 2 + ELT * 4 + (size_t)kM * 384 * 4);
  int* nbr = (int*)(wimg + 21 * SLAB);
  int* cnt = nbr + kN * MAXNB;

  const u16* AN = nullptr;                  // unused A slots
  u16* BN = nullptr;                        // no bf16 output
  float* CN = nullptr;                      // no f32 output
  const float* FN = nullptr;                // no bias
  dim3 blk(256), g1(729, 1), g4(729, 4);
  const int W96 = 96 * 96;
  const int PG = (SLAB + 255) / 256;        // prep grid

  // ---- one-time prep: weight images, encoder, state init, neighbor lists ----
  // slabs: 0-2 cr, 3-5 dr, 6-8 gm1(x,mc,md), 9-10 gm2, 11-18 lstm(4 grp x 2), 19-20 head
  prep_w_k<<<PG, blk, 0, stream>>>(cr_W + 0*W96, 1, 0, wimg + 0*SLAB);
  prep_w_k<<<PG, blk, 0, stream>>>(cr_W + 1*W96, 1, 0, wimg + 1*SLAB);
  prep_w_k<<<PG, blk, 0, stream>>>(cr_W + 2*W96, 1, 0, wimg + 2*SLAB);
  prep_w_k<<<PG, blk, 0, stream>>>(dr_W + 0*W96, 1, 0, wimg + 3*SLAB);
  prep_w_k<<<PG, blk, 0, stream>>>(dr_W + 1*W96, 1, 0, wimg + 4*SLAB);
  prep_w_k<<<PG, blk, 0, stream>>>(dr_W + 2*W96, 1, 0, wimg + 5*SLAB);
  prep_w_k<<<PG, blk, 0, stream>>>(gm_W1 + 0*W96, 1, 0, wimg + 6*SLAB);
  prep_w_k<<<PG, blk, 0, stream>>>(gm_W1 + 1*W96, 1, 0, wimg + 7*SLAB);
  prep_w_k<<<PG, blk, 0, stream>>>(gm_W1 + 2*W96, 1, 0, wimg + 8*SLAB);
  prep_w_k<<<PG, blk, 0, stream>>>(gm_W2 + 0*W96, 1, 0, wimg + 9*SLAB);
  prep_w_k<<<PG, blk, 0, stream>>>(gm_W2 + 1*W96, 1, 0, wimg + 10*SLAB);
  for (int g = 0; g < 4; ++g) {             // LSTM col groups: {Wih, Whh} rows g*96..
    prep_w_k<<<PG, blk, 0, stream>>>(Wih, 0, 96 * g, wimg + (11 + 2*g) * SLAB);
    prep_w_k<<<PG, blk, 0, stream>>>(Whh, 0, 96 * g, wimg + (12 + 2*g) * SLAB);
  }
  prep_w_k<<<PG, blk, 0, stream>>>(r_W12 + 0*W96, 1, 0, wimg + 19*SLAB);
  prep_w_k<<<PG, blk, 0, stream>>>(r_W12 + 1*W96, 1, 0, wimg + 20*SLAB);

  encode_k<<<(kM * kH + 255) / 256, blk, 0, stream>>>(inputs, enc_W1, enc_b1, enc_W2, enc_b2, X);
  hipMemcpyAsync(h, X, ELT * sizeof(u16), hipMemcpyDeviceToDevice, stream);   // h0 = X
  hipMemcpyAsync(c, c0, ELT * sizeof(float), hipMemcpyDeviceToDevice, stream);
  build_nbr_k<<<1, 128, 0, stream>>>(adj_digit, nbr, cnt);

  for (int it = 0; it < iters; ++it) {
    // --- cell-relation MLP (3 layers) + cell message ---
    gemm96_k<1,1,0,1><<<g1, blk, 0, stream>>>(h,  AN, AN, wimg + 0*SLAB, cr_b + 0,   CN, u0, 96);
    gemm96_k<1,1,0,1><<<g1, blk, 0, stream>>>(u0, AN, AN, wimg + 1*SLAB, cr_b + 96,  CN, u1, 96);
    gemm96_k<1,0,0,1><<<g1, blk, 0, stream>>>(u1, AN, AN, wimg + 2*SLAB, cr_b + 192, CN, u2, 96);
    msg_cell_k<<<(kB * kN * (kH/4) + 255) / 256, blk, 0, stream>>>(u2, adj_cell, u0);
    // --- digit-relation MLP (3 layers) + digit message ---
    gemm96_k<1,1,0,1><<<g1, blk, 0, stream>>>(h,  AN, AN, wimg + 3*SLAB, dr_b + 0,   CN, u1, 96);
    gemm96_k<1,1,0,1><<<g1, blk, 0, stream>>>(u1, AN, AN, wimg + 4*SLAB, dr_b + 96,  CN, u2, 96);
    gemm96_k<1,0,0,1><<<g1, blk, 0, stream>>>(u2, AN, AN, wimg + 5*SLAB, dr_b + 192, CN, u1, 96);
    msg_digit_k<<<(kB * kN * kD * (kH/4) + 255) / 256, blk, 0, stream>>>(u1, nbr, cnt, u2);
    // --- gate MLP: K=288 split over {X, msg_cell(u0), msg_digit(u2)} ---
    gemm96_k<3,1,0,1><<<g1, blk, 0, stream>>>(X,  u0, u2, wimg + 6*SLAB, gm_b1, CN, u1, 96);
    gemm96_k<1,1,0,1><<<g1, blk, 0, stream>>>(u1, AN, AN, wimg + 9*SLAB,  gm_b2 + 0,  CN, u0, 96);
    gemm96_k<1,0,0,1><<<g1, blk, 0, stream>>>(u0, AN, AN, wimg + 10*SLAB, gm_b2 + 96, CN, u2, 96);
    // --- LSTM: gates(f32) = g@Wih^T + h@Whh^T; 4 column groups of 96 ---
    gemm96_k<2,0,1,0><<<g4, blk, 0, stream>>>(u2, h, AN, wimg + 11*SLAB, FN, gates, BN, 384);
    lstm_k<<<(kM * kH + 255) / 256, blk, 0, stream>>>(gates, bih, bhh, h, c);
    // --- readout head ---
    gemm96_k<1,1,0,1><<<g1, blk, 0, stream>>>(h,  AN, AN, wimg + 19*SLAB, r_b12 + 0,  CN, u0, 96);
    gemm96_k<1,1,0,1><<<g1, blk, 0, stream>>>(u0, AN, AN, wimg + 20*SLAB, r_b12 + 96, CN, u1, 96);
    head_k<<<(kM + 127) / 128, dim3(128), 0, stream>>>(u1, r_W3, r_b3, out + (size_t)it * kM);
  }
}
